// TTRModel_13039520711182
// MI455X (gfx1250) — matrix operand order repaired and checked by reading, not compiled, not run on hardware
//
#include <hip/hip_runtime.h>
#include <hip/hip_bf16.h>

typedef __attribute__((ext_vector_type(16))) __bf16 v16bf;
typedef __attribute__((ext_vector_type(8)))  float  v8f;

#define HD 128

// ---------------- utility ----------------
__global__ void zero_f32(float* __restrict__ p, int n) {
  int i = blockIdx.x * blockDim.x + threadIdx.x;
  if (i < n) p[i] = 0.0f;
}

// deg[dst] += 1 for every edge (self-loop +1 folded into rsqrt kernel)
__global__ void deg_kernel(const int* __restrict__ ei, float* __restrict__ deg, int E) {
  int e = blockIdx.x * blockDim.x + threadIdx.x;
  if (e < E) atomicAdd(&deg[ei[E + e]], 1.0f);
}

__global__ void rsqrt_kernel(float* __restrict__ d, int n) {
  int i = blockIdx.x * blockDim.x + threadIdx.x;
  if (i < n) d[i] = rsqrtf(d[i] + 1.0f);
}

// ---------------- WMMA GEMM:  out[nrows,128] = (relu?)(A[nrows,128]) @ W[128,128] (+bias) ---------
template <int RELU_IN, int HAS_BIAS>
__global__ __launch_bounds__(256) void gemm128(const float* __restrict__ A,
                                               const float* __restrict__ W,
                                               const float* __restrict__ bias,
                                               float* __restrict__ out,
                                               int nrows) {
  // W transposed into LDS as bf16: ldsW[n*128 + k]
  __shared__ __bf16 ldsW[HD * HD];
  for (int i = threadIdx.x; i < HD * HD; i += 256) {
    int k = i >> 7, n = i & 127;
    ldsW[n * HD + k] = (__bf16)W[i];
  }
  __syncthreads();

  const int wave  = threadIdx.x >> 5;   // 0..7 -> row tile within block
  const int lane  = threadIdx.x & 31;
  const int mlane = lane & 15;
  const bool hi   = lane >= 16;
  const int rowBase = blockIdx.x * 128 + wave * 16;
  const int arow    = rowBase + mlane;

  v8f c[8];
#pragma unroll
  for (int i = 0; i < 8; ++i) c[i] = (v8f){0.f,0.f,0.f,0.f,0.f,0.f,0.f,0.f};

#pragma unroll
  for (int kt = 0; kt < 4; ++kt) {            // K = 128 in steps of 32
    // A fragment, contiguous K order matching B:
    //  lanes 0-15 : elems 0..15 = K kt*32+0..15
    //  lanes 16-31: elems 0..15 = K kt*32+16..31
    v16bf a = {};
    if (arow < nrows) {
      const float* ap = A + (size_t)arow * HD + kt * 32 + (hi ? 16 : 0);
      float av[16];
#pragma unroll
      for (int j = 0; j < 8; ++j) av[j]     = ap[j];
#pragma unroll
      for (int j = 0; j < 8; ++j) av[8 + j] = ap[8 + j];
#pragma unroll
      for (int j = 0; j < 16; ++j) {
        float f = av[j];
        if (RELU_IN) f = (f > 0.0f) ? f : 0.0f;
        a[j] = (__bf16)f;
      }
    }
#pragma unroll
    for (int nt = 0; nt < 8; ++nt) {          // 8 column tiles, A reused
      // B fragment: lane -> column n, 16 consecutive K values per lane half
      const v16bf b = *reinterpret_cast<const v16bf*>(
          &ldsW[(nt * 16 + mlane) * HD + kt * 32 + (hi ? 16 : 0)]);
      c[nt] = __builtin_amdgcn_wmma_f32_16x16x32_bf16(
          false, a, false, b, (short)0, c[nt], false, false);
    }
  }

  // C layout: VGPR r -> M = r (lanes 0-15) or M = r+8 (lanes 16-31); N = mlane
#pragma unroll
  for (int nt = 0; nt < 8; ++nt) {
    const int col = nt * 16 + mlane;
    const float bb = HAS_BIAS ? bias[col] : 0.0f;
#pragma unroll
    for (int r = 0; r < 8; ++r) {
      const int row = rowBase + (hi ? 8 : 0) + r;
      if (row < nrows) out[(size_t)row * HD + col] = c[nt][r] + bb;
    }
  }
}

// ---------------- agg init: self-loop term + conv bias ----------------
__global__ void self_init(const float* __restrict__ hw, const float* __restrict__ dis,
                          const float* __restrict__ bias, float* __restrict__ agg, int nelem) {
  int i = blockIdx.x * blockDim.x + threadIdx.x;
  if (i < nelem) {
    int node = i >> 7, f = i & 127;
    float s = dis[node];
    agg[i] = hw[i] * s * s + bias[f];
  }
}

// ---------------- edge scatter: wave-per-edge, coalesced f32 atomics -------------
__global__ void edge_scatter(const int* __restrict__ ei, const float* __restrict__ hw,
                             const float* __restrict__ dis, float* __restrict__ agg, int E) {
  int t = blockIdx.x * blockDim.x + threadIdx.x;
  int e = t >> 5;
  if (e >= E) return;
  int lane = t & 31;
  int s = ei[e], d = ei[E + e];
  float coef = dis[s] * dis[d];
  const float* hs = hw + (size_t)s * HD;
  float* ad = agg + (size_t)d * HD;
#pragma unroll
  for (int j = 0; j < 4; ++j) {
    int f = lane + j * 32;
    atomicAdd(&ad[f], hs[f] * coef);
  }
}

// ---------------- mean pool over relu(h3) ----------------
__global__ void pool_kernel(const float* __restrict__ h, float* __restrict__ gemb, int nrows) {
  int f = threadIdx.x;   // 128 threads
  float acc = 0.0f;
  for (int r = blockIdx.x; r < nrows; r += gridDim.x) {
    float v = h[(size_t)r * HD + f];
    acc += (v > 0.0f) ? v : 0.0f;
  }
  atomicAdd(&gemb[f], acc);
}

// ---------------- tiny MLP heads, one block ----------------
__global__ void heads_kernel(const float* __restrict__ gemb, const float* __restrict__ ps,
                             const float* __restrict__ Wp1, const float* __restrict__ bp1,
                             const float* __restrict__ Wp2, const float* __restrict__ bp2,
                             const float* __restrict__ Wq1, const float* __restrict__ bq1,
                             const float* __restrict__ Wq2, const float* __restrict__ bq2,
                             const float* __restrict__ Wv1, const float* __restrict__ bv1,
                             const float* __restrict__ Wv2, const float* __restrict__ bv2,
                             float* __restrict__ out, float invN) {
  __shared__ float sps[256], sp1[128], sp[128], scomb[256], sq1[128], sv1[128];
  const int t = threadIdx.x;           // 256 threads
  sps[t] = ps[t];
  __syncthreads();
  if (t < 128) {                        // p1 = relu(ps @ Wp1 + bp1)
    float acc = bp1[t];
    for (int k = 0; k < 256; ++k) acc += sps[k] * Wp1[k * 128 + t];
    sp1[t] = (acc > 0.0f) ? acc : 0.0f;
  }
  __syncthreads();
  if (t < 128) {                        // p = p1 @ Wp2 + bp2
    float acc = bp2[t];
    for (int k = 0; k < 128; ++k) acc += sp1[k] * Wp2[k * 128 + t];
    sp[t] = acc;
  }
  __syncthreads();
  scomb[t] = (t < 128) ? gemb[t] * invN : sp[t - 128];
  __syncthreads();
  if (t < 128) {                        // q1 = relu(combined @ Wq1 + bq1)
    float acc = bq1[t];
    for (int k = 0; k < 256; ++k) acc += scomb[k] * Wq1[k * 128 + t];
    sq1[t] = (acc > 0.0f) ? acc : 0.0f;
  } else {                              // v1 = relu(combined @ Wv1 + bv1)
    int u = t - 128;
    float acc = bv1[u];
    for (int k = 0; k < 256; ++k) acc += scomb[k] * Wv1[k * 128 + u];
    sv1[u] = (acc > 0.0f) ? acc : 0.0f;
  }
  __syncthreads();
  if (t < 100) {                        // policy logits
    float acc = bq2[t];
    for (int k = 0; k < 128; ++k) acc += sq1[k] * Wq2[k * 100 + t];
    out[t] = acc;
  }
  if (t == 255) {                       // value
    float acc = bv2[0];
    for (int k = 0; k < 128; ++k) acc += sv1[k] * Wv2[k];
    out[100] = acc;
  }
}

// ---------------- host orchestration ----------------
extern "C" void kernel_launch(void* const* d_in, const int* in_sizes, int n_in,
                              void* d_out, int out_size, void* d_ws, size_t ws_size,
                              hipStream_t stream) {
  const float* x     = (const float*)d_in[0];
  const int*   ei    = (const int*)d_in[1];      // [2,E]: src=ei[e], dst=ei[E+e]
  const float* ps    = (const float*)d_in[2];
  const float* W_enc = (const float*)d_in[3];
  const float* b_enc = (const float*)d_in[4];
  const float* W1    = (const float*)d_in[5];
  const float* b1    = (const float*)d_in[6];
  const float* W2    = (const float*)d_in[7];
  const float* b2    = (const float*)d_in[8];
  const float* W3    = (const float*)d_in[9];
  const float* b3    = (const float*)d_in[10];
  const float* Wp1   = (const float*)d_in[11];
  const float* bp1   = (const float*)d_in[12];
  const float* Wp2   = (const float*)d_in[13];
  const float* bp2   = (const float*)d_in[14];
  const float* Wq1   = (const float*)d_in[15];
  const float* bq1   = (const float*)d_in[16];
  const float* Wq2   = (const float*)d_in[17];
  const float* bq2   = (const float*)d_in[18];
  const float* Wv1   = (const float*)d_in[19];
  const float* bv1   = (const float*)d_in[20];
  const float* Wv2   = (const float*)d_in[21];
  const float* bv2   = (const float*)d_in[22];
  float* out = (float*)d_out;

  const int N = in_sizes[0] / HD;
  const int E = in_sizes[1] / 2;

  // workspace: P, Q (N*128 f32 ping-pong), dis (N), gemb (128)
  char* ws = (char*)d_ws;
  const size_t bigBytes = (size_t)N * HD * sizeof(float);
  float* P    = (float*)ws;
  float* Q    = (float*)(ws + bigBytes);
  float* dis  = (float*)(ws + 2 * bigBytes);
  float* gemb = dis + N;

  const int nelem    = N * HD;
  const int eblocks  = (E + 255) / 256;
  const int nblocks  = (N + 255) / 256;
  const int feblocks = (nelem + 255) / 256;
  const int gblocks  = (N + 127) / 128;
  const int sblocks  = (int)(((size_t)E * 32 + 255) / 256);

  // degrees -> dis = rsqrt(deg+1)
  zero_f32<<<nblocks, 256, 0, stream>>>(dis, N);
  zero_f32<<<1, 128, 0, stream>>>(gemb, 128);
  deg_kernel<<<eblocks, 256, 0, stream>>>(ei, dis, E);
  rsqrt_kernel<<<nblocks, 256, 0, stream>>>(dis, N);

  // encoder: P = x @ W_enc + b_enc
  gemm128<0, 1><<<gblocks, 256, 0, stream>>>(x, W_enc, b_enc, P, N);

  // conv1: hw=Q, agg=P (raw, relu folded into next reader)
  gemm128<0, 0><<<gblocks, 256, 0, stream>>>(P, W1, nullptr, Q, N);
  self_init<<<feblocks, 256, 0, stream>>>(Q, dis, b1, P, nelem);
  edge_scatter<<<sblocks, 256, 0, stream>>>(ei, Q, dis, P, E);

  // conv2
  gemm128<1, 0><<<gblocks, 256, 0, stream>>>(P, W2, nullptr, Q, N);
  self_init<<<feblocks, 256, 0, stream>>>(Q, dis, b2, P, nelem);
  edge_scatter<<<sblocks, 256, 0, stream>>>(ei, Q, dis, P, E);

  // conv3
  gemm128<1, 0><<<gblocks, 256, 0, stream>>>(P, W3, nullptr, Q, N);
  self_init<<<feblocks, 256, 0, stream>>>(Q, dis, b3, P, nelem);
  edge_scatter<<<sblocks, 256, 0, stream>>>(ei, Q, dis, P, E);

  // graph embedding = mean(relu(h3))
  pool_kernel<<<512, 128, 0, stream>>>(P, gemb, N);

  // heads
  heads_kernel<<<1, 256, 0, stream>>>(gemb, ps, Wp1, bp1, Wp2, bp2,
                                      Wq1, bq1, Wq2, bq2, Wv1, bv1, Wv2, bv2,
                                      out, 1.0f / (float)N);
}
